// KANvolution_80041010528597
// MI455X (gfx1250) — matrix operand order repaired and checked by reading, not compiled, not run on hardware
//
#include <hip/hip_runtime.h>

// CDNA5 / gfx1250 implicit-GEMM 3x3 conv (KANvolution).
// - v_wmma_f32_16x16x32_bf16 (f32 accumulate) for the K=144 reduction
// - global_load_async_to_lds_b128 (ASYNCcnt) stages the shared input halo
// - LDS-staged output -> coalesced global_store_b128 (full cachelines)

typedef __attribute__((ext_vector_type(16))) __bf16        v16bf;
typedef __attribute__((ext_vector_type(8)))  float         v8f;
typedef __attribute__((ext_vector_type(8)))  unsigned int  v8u;

#define IMG_H   384
#define IMG_W   384
#define C_IN    16
#define N_F     32
#define BATCH   16
#define WTILES  (IMG_W / 16)        // 24 w-tiles per row
#define GROUPS  (WTILES / 8)        // 3 blocks per row (8 waves/block)
#define HALO_W  130                 // 8*16 + 2 halo columns
#define IN_SITES (3 * HALO_W)       // 390 (row, col) sites, 64 B each
#define IN_XFERS (IN_SITES * 4)     // 1560 b128 transfers

// RNE pack of two fp32 into one dword of two bf16 (prep kernel only).
__device__ __forceinline__ unsigned int pack2bf(float lo, float hi) {
    unsigned int a = __float_as_uint(lo);
    unsigned int b = __float_as_uint(hi);
    a += 0x7FFFu + ((a >> 16) & 1u);
    b += 0x7FFFu + ((b >> 16) & 1u);
    return (a >> 16) | (b & 0xFFFF0000u);
}

// ---------------------------------------------------------------------------
// Prep: B-matrix WMMA fragments of K_eff[t][c][f] = w_spline + w_silu.
// wfrag[((p*2 + nt)*32 + lane)*8 + j]: lane = 16*half + n holds column
// f = nt*16 + n, s = {2j,2j+1} -> tap t = 2p+(j>>2), c = 8*half + 2*(j&3) + {0,1}.
// Pair 4's second tap (t=9) is zero padding.
// ---------------------------------------------------------------------------
__global__ void kan_prep_weights(const float* __restrict__ wsp,
                                 const float* __restrict__ wsi,
                                 unsigned int* __restrict__ wfrag) {
    const int idx = blockIdx.x * blockDim.x + threadIdx.x;  // 0..2559
    if (idx >= 5 * 2 * 32 * 8) return;
    const int j    = idx & 7;
    const int lane = (idx >> 3) & 31;
    const int nt   = (idx >> 8) & 1;
    const int p    = idx >> 9;
    const int half = lane >> 4;
    const int n    = lane & 15;
    const int t    = 2 * p + (j >> 2);
    const int f    = nt * 16 + n;
    float lo = 0.0f, hi = 0.0f;
    if (t < 9) {
        const int kh = t / 3, kw = t % 3;
        const int i0 = ((f * C_IN + (8 * half + 2 * (j & 3) + 0)) * 3 + kh) * 3 + kw;
        const int i1 = ((f * C_IN + (8 * half + 2 * (j & 3) + 1)) * 3 + kh) * 3 + kw;
        lo = wsp[i0] + wsi[i0];
        hi = wsp[i1] + wsi[i1];
    }
    wfrag[idx] = pack2bf(lo, hi);
}

// ---------------------------------------------------------------------------
// Main kernel: block = 8 waves = 8 consecutive 16(w) x 32(f) tiles of one
// output row (b, h, W0..W0+127). Shared input halo (3 x 130 x 16 fp32,
// 24.9 KB) is staged into LDS once per block via async global->LDS b128.
// ---------------------------------------------------------------------------
__global__ __launch_bounds__(256) void kan_conv_wmma(
    const float* __restrict__ x,
    const unsigned int* __restrict__ wfrag,
    const float* __restrict__ bias,
    float* __restrict__ out) {

    __shared__ __align__(16) float smem_in[IN_SITES * C_IN];   // 24,960 B
    __shared__ __align__(16) float smem_out[8 * 16 * N_F];     // 16,384 B

    const int lane = threadIdx.x & 31;
    const int wave = threadIdx.x >> 5;

    const int grp = blockIdx.x % GROUPS;            // third of the row
    const int h   = (blockIdx.x / GROUPS) % IMG_H;
    const int bat =  blockIdx.x / (GROUPS * IMG_H);
    const int W0  = grp * 128;                      // block's first output col

    // ---- Stage input halo: rows h-1..h+1, cols W0-1..W0+128, all 16 ch ----
    {
        const unsigned lds_base = (unsigned)(size_t)(void*)smem_in;
        for (int i = threadIdx.x; i < IN_XFERS; i += 256) {
            const int chunk = i & 3;                 // 16 B chunk within site
            const int site  = i >> 2;
            const int r     = site / HALO_W;         // 0..2 -> h-1+r
            const int wc    = site - r * HALO_W;     // 0..129 -> W0-1+wc
            const int hh    = h + r - 1;
            const int ww    = W0 + wc - 1;
            if ((unsigned)hh < IMG_H && (unsigned)ww < IMG_W) {
                const unsigned long long gaddr = (unsigned long long)(const void*)(
                    x + ((((size_t)bat * IMG_H + hh) * IMG_W + ww) * C_IN + chunk * 4));
                const unsigned ldsoff = lds_base + (unsigned)i * 16u;
                asm volatile("global_load_async_to_lds_b128 %0, %1, off"
                             :: "v"(ldsoff), "v"(gaddr) : "memory");
            }
        }
        asm volatile("s_wait_asynccnt 0x0" ::: "memory");
    }
    __syncthreads();

    // ---- Per-wave 16x32 tile ----
    const int w0   = W0 + wave * 16;
    const int half = lane >> 4;
    const int m    = lane & 15;
    const int c0   = 8 * half;

    v8f acc0 = {};
    v8f acc1 = {};

#pragma unroll
    for (int p = 0; p < 5; ++p) {
        // B fragments (10 KB table, L2-resident, coalesced 2 x b128/lane)
        const v8u bw0 = *(const v8u*)(wfrag + (((p * 2 + 0) * 32) + lane) * 8);
        const v8u bw1 = *(const v8u*)(wfrag + (((p * 2 + 1) * 32) + lane) * 8);

        // A fragment from LDS: taps 2p, 2p+1 (pair 4: second half zero)
        v16bf a = {};
        {
            const int t  = 2 * p;
            const int kh = t / 3, kw = t % 3;
            if ((unsigned)(h + kh - 1) < IMG_H &&
                (unsigned)(w0 + m + kw - 1) < IMG_W) {
                const float* sp = smem_in +
                    ((kh * HALO_W) + (wave * 16 + m + kw)) * C_IN + c0;
                const float4 x0 = *(const float4*)(sp + 0);
                const float4 x1 = *(const float4*)(sp + 4);
                a[0] = (__bf16)x0.x; a[1] = (__bf16)x0.y;
                a[2] = (__bf16)x0.z; a[3] = (__bf16)x0.w;
                a[4] = (__bf16)x1.x; a[5] = (__bf16)x1.y;
                a[6] = (__bf16)x1.z; a[7] = (__bf16)x1.w;
            }
        }
        if (p < 4) {
            const int t  = 2 * p + 1;
            const int kh = t / 3, kw = t % 3;
            if ((unsigned)(h + kh - 1) < IMG_H &&
                (unsigned)(w0 + m + kw - 1) < IMG_W) {
                const float* sp = smem_in +
                    ((kh * HALO_W) + (wave * 16 + m + kw)) * C_IN + c0;
                const float4 x0 = *(const float4*)(sp + 0);
                const float4 x1 = *(const float4*)(sp + 4);
                a[8]  = (__bf16)x0.x; a[9]  = (__bf16)x0.y;
                a[10] = (__bf16)x0.z; a[11] = (__bf16)x0.w;
                a[12] = (__bf16)x1.x; a[13] = (__bf16)x1.y;
                a[14] = (__bf16)x1.z; a[15] = (__bf16)x1.w;
            }
        }

        const v16bf b0 = __builtin_bit_cast(v16bf, bw0);
        const v16bf b1 = __builtin_bit_cast(v16bf, bw1);

        acc0 = __builtin_amdgcn_wmma_f32_16x16x32_bf16(
            false, a, false, b0, (short)0, acc0, false, false);
        acc1 = __builtin_amdgcn_wmma_f32_16x16x32_bf16(
            false, a, false, b1, (short)0, acc1, false, false);
    }

    // ---- Store via LDS transpose -> 4 x global_store_b128 per lane ----
    // C/D layout: lane n = f column, VGPR r -> w row M = 8*half + r.
    const int n = lane & 15;
    const float bs0 = bias[n];
    const float bs1 = bias[16 + n];
    float* so = smem_out + wave * (16 * N_F);
#pragma unroll
    for (int r = 0; r < 8; ++r) {
        so[(8 * half + r) * N_F + n]      = acc0[r] + bs0;
        so[(8 * half + r) * N_F + 16 + n] = acc1[r] + bs1;
    }
    // Wave-private region: same-wave DS ordering (compiler inserts dscnt wait).
    float* op = out + (((size_t)bat * IMG_H + h) * IMG_W + w0) * N_F;
#pragma unroll
    for (int k = 0; k < 4; ++k) {
        const float4 v = *(const float4*)(so + lane * 16 + k * 4);
        *(float4*)(op + lane * 16 + k * 4) = v;
    }
}

// ---------------------------------------------------------------------------
extern "C" void kernel_launch(void* const* d_in, const int* in_sizes, int n_in,
                              void* d_out, int out_size, void* d_ws, size_t ws_size,
                              hipStream_t stream) {
    const float* x    = (const float*)d_in[0];
    const float* wsp  = (const float*)d_in[1];
    const float* wsi  = (const float*)d_in[2];
    const float* bias = (const float*)d_in[3];
    unsigned int* wfrag = (unsigned int*)d_ws;   // 5*2*32*8 dwords = 10,240 B

    kan_prep_weights<<<10, 256, 0, stream>>>(wsp, wsi, wfrag);

    const int blocks = BATCH * IMG_H * GROUPS;   // 18,432 blocks x 8 waves
    kan_conv_wmma<<<blocks, 256, 0, stream>>>(x, wfrag, bias, (float*)d_out);
}
